// LOCATE_10831907521148
// MI455X (gfx1250) — matrix operand order repaired and checked by reading, not compiled, not run on hardware
//
#include <hip/hip_runtime.h>
#include <hip/hip_bf16.h>

typedef __attribute__((ext_vector_type(16))) _Float16 v16h;
typedef __attribute__((ext_vector_type(8)))  _Float16 v8h;
typedef __attribute__((ext_vector_type(8)))  float    v8f;

#define NB   64      // batch
#define NF   32      // frames
#define NBOX 36      // boxes
#define REG  1024
#define HID  1024
#define ATT  1024
#define F2   3072

static __device__ __forceinline__ v16h cat8(v8h a, v8h b) {
  return __builtin_shufflevector(a, b, 0,1,2,3,4,5,6,7,8,9,10,11,12,13,14,15);
}

static __device__ __forceinline__ v8f wmma_f16(v16h a, v16h b, v8f c) {
  return __builtin_amdgcn_wmma_f32_16x16x32_f16(false, a, false, b, (short)0, c,
                                                false, false);
}

// ---------------------------------------------------------------- prep kernels
__global__ __launch_bounds__(256)
void k_cvt_f16(const float* __restrict__ src, _Float16* __restrict__ dst, int n) {
  int i = (blockIdx.x * 256 + threadIdx.x) * 4;
  if (i < n) {
    float4 v = *(const float4*)(src + i);
    dst[i + 0] = (_Float16)v.x;
    dst[i + 1] = (_Float16)v.y;
    dst[i + 2] = (_Float16)v.z;
    dst[i + 3] = (_Float16)v.w;
  }
}

// frame_feats [B,F,2048] f32 -> feat16[b,f,1024:3072] f16
__global__ __launch_bounds__(256)
void k_frame16(const float* __restrict__ frame, _Float16* __restrict__ feat16) {
  int i = (blockIdx.x * 256 + threadIdx.x) * 4;
  if (i < NB * NF * 2048) {
    int bf = i >> 11;
    int j  = i & 2047;
    float4 v = *(const float4*)(frame + i);
    _Float16* d = feat16 + (size_t)bf * F2 + 1024 + j;
    d[0] = (_Float16)v.x; d[1] = (_Float16)v.y;
    d[2] = (_Float16)v.z; d[3] = (_Float16)v.w;
  }
}

// hs_tot[b,a] = hidden[b]·s_wh_w[a] + s_wh_b[a] + s_wv_b[a]
// ht_tot[b,a] = hidden[b]·t_wh_w[a] + t_wh_b[a] + t_wv_b[a]
__global__ __launch_bounds__(256)
void k_hproj(const float* __restrict__ hidden,
             const float* __restrict__ swh, const float* __restrict__ swhb,
             const float* __restrict__ swvb,
             const float* __restrict__ twh, const float* __restrict__ twhb,
             const float* __restrict__ twvb,
             float* __restrict__ hs_tot, float* __restrict__ ht_tot) {
  int idx = blockIdx.x * 256 + threadIdx.x;      // 0 .. 131071
  int which = idx >> 16;
  int a = idx & 1023;
  int b = (idx >> 10) & 63;
  const float* w = (which ? twh : swh) + (size_t)a * 1024;
  const float* h = hidden + (size_t)b * 1024;
  float acc = 0.f;
  for (int k = 0; k < 1024; k += 4) {
    float4 hv = *(const float4*)(h + k);
    float4 wv = *(const float4*)(w + k);
    acc += hv.x * wv.x + hv.y * wv.y + hv.z * wv.z + hv.w * wv.w;
  }
  if (which == 0) hs_tot[b * 1024 + a] = acc + swhb[a] + swvb[a];
  else            ht_tot[b * 1024 + a] = acc + twhb[a] + twvb[a];
}

// ------------------------------------------------------------ spatial attention
// one workgroup per (b,f); 8 waves; M = 48 padded boxes, N = ATT (64 tiles), K = 1024
#define SLDA (1024 + 8)   // halves; lane row-stride 516 dwords == 4 mod 64 banks
__global__ __launch_bounds__(256)
void k_spatial(const float* __restrict__ obj,       // [B,F,36,1024]
               const _Float16* __restrict__ wv16,   // [1024(a),1024(k)]
               const float* __restrict__ hs_tot,    // [B,1024]
               const float* __restrict__ wa,        // [1024]
               float* __restrict__ obj_att,         // [B,F,1024]
               _Float16* __restrict__ feat16) {     // [B,F,3072]
  __shared__ __align__(32) _Float16 lA[48 * SLDA];
  __shared__ float lscore[48];
  __shared__ float lalpha[48];

  const int bf  = blockIdx.x;
  const int b   = bf >> 5;
  const int tid = threadIdx.x;
  const float* src = obj + (size_t)bf * (NBOX * 1024);

  if (tid < 48) lscore[tid] = 0.f;

  // stage object tile as f16 (rows 36..47 zero-padded)
  for (int i = tid; i < 48 * 1024 / 4; i += 256) {
    int e = i * 4, row = e >> 10, col = e & 1023;
    _Float16* d = &lA[row * SLDA + col];
    if (row < NBOX) {
      float4 v = *(const float4*)(src + row * 1024 + col);
      d[0] = (_Float16)v.x; d[1] = (_Float16)v.y;
      d[2] = (_Float16)v.z; d[3] = (_Float16)v.w;
    } else {
      d[0] = d[1] = d[2] = d[3] = (_Float16)0.f;
    }
  }
  __syncthreads();

  const int wave = tid >> 5, lane = tid & 31;
  const int lhalf = lane >> 4, lmod = lane & 15;

  float sloc[3][8];
#pragma unroll
  for (int rt = 0; rt < 3; ++rt)
#pragma unroll
    for (int r = 0; r < 8; ++r) sloc[rt][r] = 0.f;

  for (int t = 0; t < 8; ++t) {                    // 8 column tiles per wave
    const int a = (wave * 8 + t) * 16 + lmod;
    const _Float16* wcol = wv16 + (size_t)a * 1024 + lhalf * 16;
    v8f acc0 = {}, acc1 = {}, acc2 = {};
#pragma unroll 2
    for (int k = 0; k < 1024; k += 32) {
      v16h bfr = *(const v16h*)(wcol + k);         // 32B contiguous K for column a
      const _Float16* a0p = &lA[(lmod +  0) * SLDA + k + lhalf * 16];
      const _Float16* a1p = &lA[(lmod + 16) * SLDA + k + lhalf * 16];
      const _Float16* a2p = &lA[(lmod + 32) * SLDA + k + lhalf * 16];
      v16h af0 = cat8(*(const v8h*)a0p, *(const v8h*)(a0p + 8));
      v16h af1 = cat8(*(const v8h*)a1p, *(const v8h*)(a1p + 8));
      v16h af2 = cat8(*(const v8h*)a2p, *(const v8h*)(a2p + 8));
      acc0 = wmma_f16(af0, bfr, acc0);
      acc1 = wmma_f16(af1, bfr, acc1);
      acc2 = wmma_f16(af2, bfr, acc2);
    }
    const float hadd = hs_tot[b * 1024 + a];
    const float waw  = wa[a];
#pragma unroll
    for (int r = 0; r < 8; ++r) {
      sloc[0][r] += tanhf(acc0[r] + hadd) * waw;
      sloc[1][r] += tanhf(acc1[r] + hadd) * waw;
      sloc[2][r] += tanhf(acc2[r] + hadd) * waw;
    }
  }
#pragma unroll
  for (int rt = 0; rt < 3; ++rt)
#pragma unroll
    for (int r = 0; r < 8; ++r)
      atomicAdd(&lscore[rt * 16 + lhalf * 8 + r], sloc[rt][r]);
  __syncthreads();

  if (tid == 0) {                                  // softmax over 36 boxes
    float mx = -1e30f;
    for (int n = 0; n < NBOX; ++n) mx = fmaxf(mx, lscore[n]);
    float s = 0.f;
    for (int n = 0; n < NBOX; ++n) { float e = __expf(lscore[n] - mx); lalpha[n] = e; s += e; }
    float inv = 1.f / s;
    for (int n = 0; n < NBOX; ++n) lalpha[n] *= inv;
  }
  __syncthreads();

  for (int d = tid; d < 1024; d += 256) {          // alpha-weighted pooling
    float acc = 0.f;
    for (int n = 0; n < NBOX; ++n) acc += lalpha[n] * (float)lA[n * SLDA + d];
    obj_att[(size_t)bf * 1024 + d] = acc;
    feat16[(size_t)bf * F2 + d]    = (_Float16)acc;
  }
}

// ----------------------------------------------------------- temporal attention
// one workgroup per batch; M = 32 frames (2 tiles), N = HID (64 tiles), K = 3072
#define TKC  1536
#define TLDA (TKC + 8)
__global__ __launch_bounds__(256)
void k_temporal(const _Float16* __restrict__ feat16,  // [B,F,3072]
                const _Float16* __restrict__ tv16,    // [1024(a),3072(k)]
                const float* __restrict__ ht_tot,     // [B,1024]
                const float* __restrict__ twa,        // [1024]
                const float* __restrict__ obj_att,    // [B,F,1024]
                const float* __restrict__ frame,      // [B,F,2048]
                float* __restrict__ out) {            // [B,3072]
  __shared__ __align__(32) _Float16 lA[32 * TLDA];
  __shared__ float lscore[32];
  __shared__ float lbeta[32];

  const int b = blockIdx.x;
  const int tid = threadIdx.x;
  const int wave = tid >> 5, lane = tid & 31;
  const int lhalf = lane >> 4, lmod = lane & 15;

  if (tid < 32) lscore[tid] = 0.f;

  v8f acc[8][2];
#pragma unroll
  for (int t = 0; t < 8; ++t) { acc[t][0] = (v8f){}; acc[t][1] = (v8f){}; }

  for (int c = 0; c < 2; ++c) {                    // two K-chunks of 1536
    __syncthreads();
    for (int i = tid; i < 32 * (TKC / 8); i += 256) {
      int row = i / (TKC / 8), cc = (i % (TKC / 8)) * 8;
      *(v8h*)&lA[row * TLDA + cc] =
          *(const v8h*)(feat16 + (size_t)(b * NF + row) * F2 + c * TKC + cc);
    }
    __syncthreads();
#pragma unroll
    for (int t = 0; t < 8; ++t) {
      const int a = (wave * 8 + t) * 16 + lmod;
      const _Float16* wcol = tv16 + (size_t)a * F2 + c * TKC + lhalf * 16;
#pragma unroll 2
      for (int k = 0; k < TKC; k += 32) {
        v16h bfr = *(const v16h*)(wcol + k);
        const _Float16* a0p = &lA[(lmod +  0) * TLDA + k + lhalf * 16];
        const _Float16* a1p = &lA[(lmod + 16) * TLDA + k + lhalf * 16];
        v16h af0 = cat8(*(const v8h*)a0p, *(const v8h*)(a0p + 8));
        v16h af1 = cat8(*(const v8h*)a1p, *(const v8h*)(a1p + 8));
        acc[t][0] = wmma_f16(af0, bfr, acc[t][0]);
        acc[t][1] = wmma_f16(af1, bfr, acc[t][1]);
      }
    }
  }

  float sloc[2][8];
#pragma unroll
  for (int rt = 0; rt < 2; ++rt)
#pragma unroll
    for (int r = 0; r < 8; ++r) sloc[rt][r] = 0.f;
#pragma unroll
  for (int t = 0; t < 8; ++t) {
    const int a = (wave * 8 + t) * 16 + lmod;
    const float hadd = ht_tot[b * 1024 + a];
    const float w = twa[a];
#pragma unroll
    for (int r = 0; r < 8; ++r) {
      sloc[0][r] += tanhf(acc[t][0][r] + hadd) * w;
      sloc[1][r] += tanhf(acc[t][1][r] + hadd) * w;
    }
  }
#pragma unroll
  for (int rt = 0; rt < 2; ++rt)
#pragma unroll
    for (int r = 0; r < 8; ++r)
      atomicAdd(&lscore[rt * 16 + lhalf * 8 + r], sloc[rt][r]);
  __syncthreads();

  if (tid == 0) {                                  // softmax over 32 frames
    float mx = -1e30f;
    for (int f = 0; f < NF; ++f) mx = fmaxf(mx, lscore[f]);
    float s = 0.f;
    for (int f = 0; f < NF; ++f) { float e = __expf(lscore[f] - mx); lbeta[f] = e; s += e; }
    float inv = 1.f / s;
    for (int f = 0; f < NF; ++f) lbeta[f] *= inv;
  }
  __syncthreads();

  for (int d = tid; d < F2; d += 256) {            // beta-weighted sum in f32
    float a2 = 0.f;
    for (int f = 0; f < NF; ++f) {
      float v = (d < 1024)
                    ? obj_att[((size_t)b * NF + f) * 1024 + d]
                    : frame[((size_t)b * NF + f) * 2048 + (d - 1024)];
      a2 += lbeta[f] * v;
    }
    out[(size_t)b * F2 + d] = a2;
  }
}

// --------------------------------------------------------------------- launcher
extern "C" void kernel_launch(void* const* d_in, const int* in_sizes, int n_in,
                              void* d_out, int out_size, void* d_ws, size_t ws_size,
                              hipStream_t stream) {
  const float* frame  = (const float*)d_in[0];
  const float* obj    = (const float*)d_in[1];
  const float* hidden = (const float*)d_in[2];
  const float* swh_w  = (const float*)d_in[3];
  const float* swh_b  = (const float*)d_in[4];
  const float* swv_w  = (const float*)d_in[5];
  const float* swv_b  = (const float*)d_in[6];
  const float* swa    = (const float*)d_in[7];
  const float* twh_w  = (const float*)d_in[8];
  const float* twh_b  = (const float*)d_in[9];
  const float* twv_w  = (const float*)d_in[10];
  const float* twv_b  = (const float*)d_in[11];
  const float* twa    = (const float*)d_in[12];
  float* out = (float*)d_out;

  char* ws = (char*)d_ws;
  _Float16* wv16   = (_Float16*)(ws);                          //  2 MB
  _Float16* tv16   = (_Float16*)(ws + (size_t)(2  << 20));     //  6 MB
  _Float16* feat16 = (_Float16*)(ws + (size_t)(8  << 20));     // 12 MB
  float*    objatt = (float*)   (ws + (size_t)(20 << 20));     //  8 MB
  float*    hs_tot = (float*)   (ws + (size_t)(28 << 20));     // 256 KB
  float*    ht_tot = (float*)   (ws + (size_t)(28 << 20) + 64 * 1024 * 4);

  k_cvt_f16<<<(1024 * 1024 / 4 + 255) / 256, 256, 0, stream>>>(swv_w, wv16, 1024 * 1024);
  k_cvt_f16<<<(1024 * 3072 / 4 + 255) / 256, 256, 0, stream>>>(twv_w, tv16, 1024 * 3072);
  k_frame16<<<(NB * NF * 2048 / 4 + 255) / 256, 256, 0, stream>>>(frame, feat16);
  k_hproj<<<(2 * NB * 1024) / 256, 256, 0, stream>>>(hidden, swh_w, swh_b, swv_b,
                                                     twh_w, twh_b, twv_b, hs_tot, ht_tot);
  k_spatial<<<NB * NF, 256, 0, stream>>>(obj, wv16, hs_tot, swa, objatt, feat16);
  k_temporal<<<NB, 256, 0, stream>>>(feat16, tv16, ht_tot, twa, objatt, frame, out);
}
